// sLSTMBlock_37950331028117
// MI455X (gfx1250) — matrix operand order repaired and checked by reading, not compiled, not run on hardware
//
#include <hip/hip_runtime.h>

#define B_BATCH 32768
#define D_IN    512
#define H_HID   512
#define UP_PAD  704   // 682 padded to multiple of 64 (padded cols/rows are zero)

typedef __attribute__((ext_vector_type(16))) _Float16 v16h;
typedef __attribute__((ext_vector_type(8)))  _Float16 h8v;
typedef __attribute__((ext_vector_type(8)))  float    v8f;
typedef int async_v4i __attribute__((vector_size(16)));   // matches builtin param type

#define GLOBAL_AS __attribute__((address_space(1)))
#define LDS_AS    __attribute__((address_space(3)))

// ---- CDNA5 async global->LDS staging (ASYNCcnt path), with portable fallback ----
__device__ __forceinline__ void stage16(void* lds_dst, const void* gsrc) {
#if __has_builtin(__builtin_amdgcn_global_load_async_to_lds_b128)
  __builtin_amdgcn_global_load_async_to_lds_b128(
      (GLOBAL_AS async_v4i*)gsrc, (LDS_AS async_v4i*)lds_dst, 0, 0);
#else
  *(h8v*)lds_dst = *(const h8v*)gsrc;
#endif
}

__device__ __forceinline__ void stage_wait() {
#if __has_builtin(__builtin_amdgcn_global_load_async_to_lds_b128)
#if __has_builtin(__builtin_amdgcn_s_wait_asynccnt)
  __builtin_amdgcn_s_wait_asynccnt(0);
#else
  asm volatile("s_wait_asynccnt 0x0" ::: "memory");
#endif
#endif
}

__device__ __forceinline__ v16h make_frag(h8v lo, h8v hi) {
  v16h r;
#pragma unroll
  for (int i = 0; i < 8; ++i) { r[i] = lo[i]; r[i + 8] = hi[i]; }
  return r;
}

// A-matrix fragment (16x32 f16) from row-major [m][k], leading dim ld.
// Per ISA 16-bit A layout: lane m=lane&15, half=lane>>4;
// VGPR j<4 holds K=half*8+2j,+1 ; VGPR j>=4 holds K=16+half*8+2(j-4),+1
// -> two contiguous 8-element (16B) chunks at k0+half*8 and k0+16+half*8.
__device__ __forceinline__ v16h load_a_frag(const _Float16* base, int ld, int m0, int k0) {
  int lane = threadIdx.x & 31;
  const _Float16* p = base + (size_t)(m0 + (lane & 15)) * ld + k0 + ((lane >> 4) << 3);
  h8v lo = *(const h8v*)p;
  h8v hi = *(const h8v*)(p + 16);
  return make_frag(lo, hi);
}

// B-matrix fragment (32x16 f16) from N-major storage Bsrc[n][k], leading dim ld.
// Lane n=lane&15, half=lane>>4; VGPR j<4 holds K=half*8+2j,+1 ; VGPR j>=4 holds K=16+half*8+2(j-4),+1
// -> two contiguous 8-element (16B) chunks at k0+half*8 and k0+16+half*8.
__device__ __forceinline__ v16h load_b_frag(const _Float16* base, int ld, int n0, int k0) {
  int lane = threadIdx.x & 31;
  const _Float16* p = base + (size_t)(n0 + (lane & 15)) * ld + k0 + ((lane >> 4) << 3);
  h8v lo = *(const h8v*)p;
  h8v hi = *(const h8v*)(p + 16);
  return make_frag(lo, hi);
}

__device__ __forceinline__ v8f wmma_f16(v16h a, v16h b, v8f c) {
  return __builtin_amdgcn_wmma_f32_16x16x32_f16(false, a, false, b, (short)0, c, false, false);
}

// ---------------- weight packing ----------------
struct GatePack     { const float* W[4]; const float* R[4]; };
struct GateBiasPack { const float* Wb[4]; const float* Rb[4]; };

// GW[c][g][n][k] : k<256 -> Wg_w[c][n][k], k>=256 -> Rg_w[c][n][k-256]   (f16)
__global__ __launch_bounds__(256) void pack_gates_kernel(GatePack p, _Float16* __restrict__ GW) {
  int idx = blockIdx.x * 256 + threadIdx.x;   // 2*4*256*512 total
  int k = idx & 511;
  int n = (idx >> 9) & 255;
  int g = (idx >> 17) & 3;
  int c = (idx >> 19) & 1;
  const float* src = (k < 256) ? p.W[g] : p.R[g];
  GW[idx] = (_Float16)src[((c << 8) + n) * 256 + (k & 255)];
}

__global__ __launch_bounds__(256) void pack_gbias_kernel(GateBiasPack p, float* __restrict__ gb) {
  int idx = blockIdx.x * 256 + threadIdx.x;   // 2048 total
  int n = idx & 255, g = (idx >> 8) & 3, c = (idx >> 10) & 1;
  gb[idx] = p.Wb[g][(c << 8) + n] + p.Rb[g][(c << 8) + n];
}

// upw16[mat][n][k], n in [0,704), zero-padded beyond 682
__global__ __launch_bounds__(256) void pack_up_kernel(const float* __restrict__ upl,
                                                      const float* __restrict__ upr,
                                                      _Float16* __restrict__ upw) {
  int idx = blockIdx.x * 256 + threadIdx.x;   // 2*704*512 total
  int k = idx & 511;
  int n = (idx >> 9) % UP_PAD;
  int mat = idx / (UP_PAD * 512);
  const float* src = mat ? upr : upl;
  upw[idx] = (_Float16)((n < 682) ? src[n * 512 + k] : 0.f);
}

__global__ __launch_bounds__(256) void pack_upb_kernel(const float* __restrict__ bl,
                                                       const float* __restrict__ br,
                                                       float* __restrict__ ub) {
  int idx = blockIdx.x * 256 + threadIdx.x;
  if (idx >= 2 * UP_PAD) return;
  int n = idx % UP_PAD, mat = idx / UP_PAD;
  const float* s = mat ? br : bl;
  ub[idx] = (n < 682) ? s[n] : 0.f;
}

// downw16[n][k], k in [0,704), zero-padded beyond 682
__global__ __launch_bounds__(256) void pack_down_kernel(const float* __restrict__ dw,
                                                        _Float16* __restrict__ dwh) {
  int idx = blockIdx.x * 256 + threadIdx.x;   // 512*704 total
  int k = idx % UP_PAD, n = idx / UP_PAD;
  dwh[idx] = (_Float16)((k < 682) ? dw[n * 682 + k] : 0.f);
}

// ---------------- per-row prep: LayerNorm + causal conv + SiLU; f16 copies ----------------
__global__ __launch_bounds__(256) void prep_kernel(
    const float* __restrict__ x, const float* __restrict__ hprev,
    const float* __restrict__ ln_g, const float* __restrict__ ln_b,
    const float* __restrict__ conv_w, const float* __restrict__ conv_b,
    _Float16* __restrict__ x16, _Float16* __restrict__ xc16, _Float16* __restrict__ h16) {
  __shared__ float xs[512];
  __shared__ float rs[8], rq[8];
  int row = blockIdx.x, tid = threadIdx.x;
  int wave = tid >> 5, lane = tid & 31;
  size_t rbase = (size_t)row * 512;

  float2 v  = *(const float2*)(x + rbase + tid * 2);
  float2 hv = *(const float2*)(hprev + rbase + tid * 2);
  x16[rbase + tid * 2]     = (_Float16)v.x;
  x16[rbase + tid * 2 + 1] = (_Float16)v.y;
  h16[rbase + tid * 2]     = (_Float16)hv.x;
  h16[rbase + tid * 2 + 1] = (_Float16)hv.y;

  float s = v.x + v.y, q = v.x * v.x + v.y * v.y;
#pragma unroll
  for (int o = 16; o > 0; o >>= 1) { s += __shfl_down(s, o, 32); q += __shfl_down(q, o, 32); }
  if (lane == 0) { rs[wave] = s; rq[wave] = q; }
  __syncthreads();
  float st = 0.f, qt = 0.f;
#pragma unroll
  for (int w = 0; w < 8; ++w) { st += rs[w]; qt += rq[w]; }
  float mean = st * (1.f / 512.f);
  float var  = qt * (1.f / 512.f) - mean * mean;
  float rstd = rsqrtf(var + 1e-5f);

  float n0 = (v.x - mean) * rstd * ln_g[tid * 2]     + ln_b[tid * 2];
  float n1 = (v.y - mean) * rstd * ln_g[tid * 2 + 1] + ln_b[tid * 2 + 1];
  xs[tid * 2] = n0;
  xs[tid * 2 + 1] = n1;
  __syncthreads();

  float cw[4];
#pragma unroll
  for (int k = 0; k < 4; ++k) cw[k] = conv_w[k];
  float cb = conv_b[0];
#pragma unroll
  for (int e = 0; e < 2; ++e) {
    int d = tid * 2 + e;
    float a = cb;
#pragma unroll
    for (int k = 0; k < 4; ++k) {
      int j = d - 3 + k;
      if (j >= 0) a += cw[k] * xs[j];
    }
    float sg = 1.f / (1.f + __expf(-a));
    xc16[rbase + d] = (_Float16)(a * sg);
  }
}

// ---------------- 4-gate block-diag WMMA GEMM + fused sLSTM state update ----------------
// grid (B/128, 4, 2): z=c (diag block), y=64-col tile within block, x=128-row tile
// Double-buffered async LDS staging: DMA of chunk kt+1 overlaps WMMA of chunk kt.
__global__ __launch_bounds__(256) void gates_kernel(
    const _Float16* __restrict__ x16, const _Float16* __restrict__ xc16,
    const _Float16* __restrict__ h16, const _Float16* __restrict__ GW,
    const float* __restrict__ gbias,
    const float* __restrict__ c_prev, const float* __restrict__ n_prev,
    const float* __restrict__ m_prev, float* __restrict__ out) {
  __shared__ __align__(16) _Float16 Bs[2][4][64][32];   // ping-pong [buf][gate][n][k], 32 KB
  const size_t BH = (size_t)B_BATCH * H_HID;
  int tid = threadIdx.x;
  int wave = tid >> 5, lane = tid & 31;
  int c  = blockIdx.z;
  int n0 = blockIdx.y * 64;
  int mw = blockIdx.x * 128 + wave * 16;

  const v8f vzero = {0.f, 0.f, 0.f, 0.f, 0.f, 0.f, 0.f, 0.f};
  v8f acc[4][4];                                     // [n-subtile][gate z,i,f,o]
#pragma unroll
  for (int s = 0; s < 4; ++s)
#pragma unroll
    for (int g = 0; g < 4; ++g) acc[s][g] = vzero;

  int sg = tid >> 6, sn = tid & 63;                  // staging: gate / n-row
  const _Float16* srow = GW + (size_t)(((c << 2) | sg) * 256 + n0 + sn) * 512;

  // prologue: stage chunk 0 into buffer 0
  {
    const _Float16* s4 = srow;
    _Float16* d4 = &Bs[0][sg][sn][0];
#pragma unroll
    for (int p = 0; p < 4; ++p) stage16(d4 + p * 8, s4 + p * 8);
  }

  for (int kt = 0; kt < 16; ++kt) {
    int cur = kt & 1;
    stage_wait();        // my async writes into Bs[cur] done
    __syncthreads();     // all waves: Bs[cur] filled; reads of Bs[cur^1] (iter kt-1) done
    if (kt + 1 < 16) {   // kick off DMA for next chunk while we compute this one
      const _Float16* s4 = srow + (kt + 1) * 32;
      _Float16* d4 = &Bs[cur ^ 1][sg][sn][0];
#pragma unroll
      for (int p = 0; p < 4; ++p) stage16(d4 + p * 8, s4 + p * 8);
    }

    v16h af0, af1;                                   // z,o use x ; i,f use x_conv ; K>=256 -> h_prev
    if (kt < 8) {
      int kx = c * 256 + kt * 32;
      af0 = load_a_frag(x16,  512, mw, kx);
      af1 = load_a_frag(xc16, 512, mw, kx);
    } else {
      int kh = c * 256 + (kt - 8) * 32;
      af0 = load_a_frag(h16, 512, mw, kh);
      af1 = af0;
    }
#pragma unroll
    for (int s = 0; s < 4; ++s) {
#pragma unroll
      for (int g = 0; g < 4; ++g) {
        v16h b = load_b_frag(&Bs[cur][g][0][0], 32, s * 16, 0);
        acc[s][g] = wmma_f16((g == 1 || g == 2) ? af1 : af0, b, acc[s][g]);
      }
    }
  }

  int halfl = lane >> 4, nl = lane & 15;
#pragma unroll
  for (int s = 0; s < 4; ++s) {
    int ncol = n0 + s * 16 + nl;
    float bz = gbias[((c << 2) + 0) * 256 + ncol];
    float bi = gbias[((c << 2) + 1) * 256 + ncol];
    float bf = gbias[((c << 2) + 2) * 256 + ncol];
    float bo = gbias[((c << 2) + 3) * 256 + ncol];
    int col = (c << 8) + ncol;
#pragma unroll
    for (int r = 0; r < 8; ++r) {
      int row = mw + halfl * 8 + r;
      size_t idx = (size_t)row * H_HID + col;
      float zp = acc[s][0][r] + bz;
      float ip = acc[s][1][r] + bi;
      float fp = acc[s][2][r] + bf;
      float op = acc[s][3][r] + bo;
      float mp = m_prev[idx], cp = c_prev[idx], np = n_prev[idx];
      float zt = tanhf(zp);
      float ot = 1.f / (1.f + __expf(-op));
      float mt = fmaxf(fp + mp, ip);
      float iv = __expf(ip - mt);
      float fv = __expf(fp + mp - mt);
      float ct = fv * cp + iv * zt;
      float nt = fv * np + iv;
      float ht = ot * (ct / nt);
      out[BH + idx]     = ht;
      out[2 * BH + idx] = ct;
      out[3 * BH + idx] = nt;
      out[4 * BH + idx] = mt;
    }
  }
}

// ---------------- GroupNorm(2, 512) -> f16 ----------------
__global__ __launch_bounds__(256) void gn_kernel(const float* __restrict__ ht,
                                                 const float* __restrict__ gn_g,
                                                 const float* __restrict__ gn_b,
                                                 _Float16* __restrict__ outn) {
  __shared__ float rs[8], rq[8];
  int row = blockIdx.x, tid = threadIdx.x;
  int wave = tid >> 5, lane = tid & 31;
  int grp = tid >> 7, j = tid & 127;
  int col = grp * 256 + j * 2;
  const float* hr = ht + (size_t)row * 512;
  float a = hr[col], b = hr[col + 1];
  float s = a + b, q = a * a + b * b;
#pragma unroll
  for (int o = 16; o > 0; o >>= 1) { s += __shfl_down(s, o, 32); q += __shfl_down(q, o, 32); }
  if (lane == 0) { rs[wave] = s; rq[wave] = q; }
  __syncthreads();
  float st = rs[grp * 4] + rs[grp * 4 + 1] + rs[grp * 4 + 2] + rs[grp * 4 + 3];
  float qt = rq[grp * 4] + rq[grp * 4 + 1] + rq[grp * 4 + 2] + rq[grp * 4 + 3];
  float mean = st * (1.f / 256.f);
  float var  = qt * (1.f / 256.f) - mean * mean;
  float rstd = rsqrtf(var + 1e-5f);
  outn[(size_t)row * 512 + col]     = (_Float16)((a - mean) * rstd * gn_g[col]     + gn_b[col]);
  outn[(size_t)row * 512 + col + 1] = (_Float16)((b - mean) * rstd * gn_g[col + 1] + gn_b[col + 1]);
}

// ---------------- MLP up: left/right GEMM + exact GELU gate -> f16 gated [B,704] ----------------
__global__ __launch_bounds__(256) void mlp_up_kernel(
    const _Float16* __restrict__ xn, const _Float16* __restrict__ upw,
    const float* __restrict__ upb, _Float16* __restrict__ gated) {
  __shared__ __align__(16) _Float16 Bs[2][2][64][32];   // ping-pong, 16 KB
  int tid = threadIdx.x;
  int wave = tid >> 5, lane = tid & 31;
  int n0 = blockIdx.y * 64;
  int mw = blockIdx.x * 128 + wave * 16;

  const v8f vzero = {0.f, 0.f, 0.f, 0.f, 0.f, 0.f, 0.f, 0.f};
  v8f acc[4][2];
#pragma unroll
  for (int s = 0; s < 4; ++s) { acc[s][0] = vzero; acc[s][1] = vzero; }

  int smat = tid >> 7, sn = (tid >> 1) & 63, spart = tid & 1;
  const _Float16* srow = upw + ((size_t)smat * UP_PAD + n0 + sn) * 512;

  {
    const _Float16* s4 = srow + spart * 16;
    _Float16* d4 = &Bs[0][smat][sn][spart * 16];
    stage16(d4, s4);
    stage16(d4 + 8, s4 + 8);
  }

  for (int kt = 0; kt < 16; ++kt) {
    int cur = kt & 1;
    stage_wait();
    __syncthreads();
    if (kt + 1 < 16) {
      const _Float16* s4 = srow + (kt + 1) * 32 + spart * 16;
      _Float16* d4 = &Bs[cur ^ 1][smat][sn][spart * 16];
      stage16(d4, s4);
      stage16(d4 + 8, s4 + 8);
    }
    v16h a = load_a_frag(xn, 512, mw, kt * 32);
#pragma unroll
    for (int s = 0; s < 4; ++s) {
      v16h b0 = load_b_frag(&Bs[cur][0][0][0], 32, s * 16, 0);
      v16h b1 = load_b_frag(&Bs[cur][1][0][0], 32, s * 16, 0);
      acc[s][0] = wmma_f16(a, b0, acc[s][0]);
      acc[s][1] = wmma_f16(a, b1, acc[s][1]);
    }
  }

  int halfl = lane >> 4, nl = lane & 15;
#pragma unroll
  for (int s = 0; s < 4; ++s) {
    int col = n0 + s * 16 + nl;
    float bl = upb[col];
    float br = upb[UP_PAD + col];
#pragma unroll
    for (int r = 0; r < 8; ++r) {
      int row = mw + halfl * 8 + r;
      float l  = acc[s][0][r] + bl;
      float rr = acc[s][1][r] + br;
      float ge = 0.5f * rr * (1.f + erff(rr * 0.70710678118f));
      gated[(size_t)row * UP_PAD + col] = (_Float16)(l * ge);
    }
  }
}

// ---------------- MLP down: final = gated @ down_w^T + down_b + x ----------------
__global__ __launch_bounds__(256) void mlp_down_kernel(
    const _Float16* __restrict__ gated, const _Float16* __restrict__ dwh,
    const float* __restrict__ down_b, const float* __restrict__ x,
    float* __restrict__ out) {
  __shared__ __align__(16) _Float16 Bs[2][64][32];   // ping-pong, 8 KB
  int tid = threadIdx.x;
  int wave = tid >> 5, lane = tid & 31;
  int n0 = blockIdx.y * 64;
  int mw = blockIdx.x * 128 + wave * 16;

  const v8f vzero = {0.f, 0.f, 0.f, 0.f, 0.f, 0.f, 0.f, 0.f};
  v8f acc[4];
#pragma unroll
  for (int s = 0; s < 4; ++s) acc[s] = vzero;

  int sn = tid >> 2, spart = tid & 3;
  const _Float16* srow = dwh + (size_t)(n0 + sn) * UP_PAD;

  stage16(&Bs[0][sn][spart * 8], srow + spart * 8);

  for (int kt = 0; kt < 22; ++kt) {           // K = 704
    int cur = kt & 1;
    stage_wait();
    __syncthreads();
    if (kt + 1 < 22)
      stage16(&Bs[cur ^ 1][sn][spart * 8], srow + (kt + 1) * 32 + spart * 8);
    v16h a = load_a_frag(gated, UP_PAD, mw, kt * 32);
#pragma unroll
    for (int s = 0; s < 4; ++s) {
      v16h b = load_b_frag(&Bs[cur][0][0], 32, s * 16, 0);
      acc[s] = wmma_f16(a, b, acc[s]);
    }
  }

  int halfl = lane >> 4, nl = lane & 15;
#pragma unroll
  for (int s = 0; s < 4; ++s) {
    int col = n0 + s * 16 + nl;
    float db = down_b[col];
#pragma unroll
    for (int r = 0; r < 8; ++r) {
      int row = mw + halfl * 8 + r;
      size_t idx = (size_t)row * 512 + col;
      out[idx] = acc[s][r] + db + x[idx];
    }
  }
}

extern "C" void kernel_launch(void* const* d_in, const int* in_sizes, int n_in,
                              void* d_out, int out_size, void* d_ws, size_t ws_size,
                              hipStream_t stream) {
  (void)in_sizes; (void)n_in; (void)out_size; (void)ws_size;
  const float* x      = (const float*)d_in[0];
  const float* h_prev = (const float*)d_in[1];
  const float* c_prev = (const float*)d_in[2];
  const float* n_prev = (const float*)d_in[3];
  const float* m_prev = (const float*)d_in[4];
  const float* ln_g   = (const float*)d_in[5];
  const float* ln_b   = (const float*)d_in[6];
  const float* conv_w = (const float*)d_in[7];
  const float* conv_b = (const float*)d_in[8];
  const float* gn_g   = (const float*)d_in[9];
  const float* gn_b   = (const float*)d_in[10];
  const float* upl_w  = (const float*)d_in[11];
  const float* upl_b  = (const float*)d_in[12];
  const float* upr_w  = (const float*)d_in[13];
  const float* upr_b  = (const float*)d_in[14];
  const float* down_w = (const float*)d_in[15];
  const float* down_b = (const float*)d_in[16];

  GatePack gp; GateBiasPack gbp;
  for (int g = 0; g < 4; ++g) {               // z, i, f, o
    gp.W[g]   = (const float*)d_in[17 + 2 * g];
    gbp.Wb[g] = (const float*)d_in[18 + 2 * g];
    gp.R[g]   = (const float*)d_in[25 + 2 * g];
    gbp.Rb[g] = (const float*)d_in[26 + 2 * g];
  }

  char* ws = (char*)d_ws;
  size_t off = 0;
  auto take = [&](size_t bytes) -> char* {
    char* p = ws + off;
    off += (bytes + 255) & ~(size_t)255;
    return p;
  };
  _Float16* x16   = (_Float16*)take((size_t)B_BATCH * 512 * 2);
  _Float16* xc16  = (_Float16*)take((size_t)B_BATCH * 512 * 2);
  _Float16* h16   = (_Float16*)take((size_t)B_BATCH * 512 * 2);
  _Float16* GW    = (_Float16*)take((size_t)2 * 4 * 256 * 512 * 2);
  float*    gbias = (float*)take((size_t)2048 * 4);
  _Float16* upw16 = (_Float16*)take((size_t)2 * UP_PAD * 512 * 2);
  float*    upb   = (float*)take((size_t)2 * UP_PAD * 4);
  _Float16* dwh   = (_Float16*)take((size_t)512 * UP_PAD * 2);
  _Float16* outn  = (_Float16*)take((size_t)B_BATCH * 512 * 2);
  _Float16* gated = (_Float16*)take((size_t)B_BATCH * UP_PAD * 2);

  pack_gates_kernel<<<4096, 256, 0, stream>>>(gp, GW);
  pack_gbias_kernel<<<8, 256, 0, stream>>>(gbp, gbias);
  pack_up_kernel<<<(2 * UP_PAD * 512) / 256, 256, 0, stream>>>(upl_w, upr_w, upw16);
  pack_upb_kernel<<<6, 256, 0, stream>>>(upl_b, upr_b, upb);
  pack_down_kernel<<<(512 * UP_PAD) / 256, 256, 0, stream>>>(down_w, dwh);
  prep_kernel<<<B_BATCH, 256, 0, stream>>>(x, h_prev, ln_g, ln_b, conv_w, conv_b,
                                           x16, xc16, h16);

  float* out = (float*)d_out;
  const size_t BH = (size_t)B_BATCH * 512;
  gates_kernel<<<dim3(B_BATCH / 128, 4, 2), 256, 0, stream>>>(
      x16, xc16, h16, GW, gbias, c_prev, n_prev, m_prev, out);
  gn_kernel<<<B_BATCH, 256, 0, stream>>>(out + BH, gn_g, gn_b, outn);
  mlp_up_kernel<<<dim3(B_BATCH / 128, UP_PAD / 64), 256, 0, stream>>>(outn, upw16, upb, gated);
  mlp_down_kernel<<<dim3(B_BATCH / 128, 512 / 64), 256, 0, stream>>>(gated, dwh, down_b, x, out);
}
